// simpleKT_47785806135763
// MI455X (gfx1250) — matrix operand order repaired and checked by reading, not compiled, not run on hardware
//
#include <hip/hip_runtime.h>

// ---------------- problem constants ----------------
#define Bz   64
#define Sz   512
#define Dz   512
#define Hz   8
#define Lz   4
#define DKz  64
#define DFFz 2048
#define NSKz 300
#define NSKp 320   // NSK padded to a multiple of 64 for full-tile B loads
#define FC1z 512
#define FC2z 256
#define Mz   (Bz * Sz)   // 32768 tokens

// ---------------- WMMA plumbing (gfx1250) ----------------
typedef __attribute__((ext_vector_type(16))) _Float16 v16h;
typedef __attribute__((ext_vector_type(8)))  float    v8f;

__device__ __forceinline__ v8f wmma_f32_f16(v16h a, v16h b, v8f c) {
  // D = A(16x32,f16) * B(32x16,f16) + C(16x16,f32)
  return __builtin_amdgcn_wmma_f32_16x16x32_f16(false, a, false, b, (short)0, c,
                                                false, false);
}

// ---------------- CDNA5 async global->LDS (ASYNCcnt) ----------------
#if defined(__HIP_DEVICE_COMPILE__) && \
    __has_builtin(__builtin_amdgcn_global_load_async_to_lds_b128)
#define HAS_ASYNC_LDS 1
#else
#define HAS_ASYNC_LDS 0
#endif

// builtin params are pointers to 16-byte int vectors (per clang diagnostic)
typedef int i4v __attribute__((vector_size(16)));
#define GPTR128(p) ((__attribute__((address_space(1))) i4v*)(p))
#define LPTR128(p) ((__attribute__((address_space(3))) i4v*)(p))

__device__ __forceinline__ void wait_async0() {
#if HAS_ASYNC_LDS
#if __has_builtin(__builtin_amdgcn_s_wait_asynccnt)
  __builtin_amdgcn_s_wait_asynccnt(0);
#else
  asm volatile("s_wait_asynccnt 0" ::: "memory");
#endif
#endif
}

// ---------------- fp32 -> fp16 convert ----------------
__global__ __launch_bounds__(256)
void cvt_k(const float* __restrict__ s, _Float16* __restrict__ d, size_t n) {
  size_t i = (size_t)blockIdx.x * 256 + threadIdx.x;
  if (i < n) d[i] = (_Float16)s[i];
}

// fp32[rows,nc] -> fp16[rows,np] zero-padded columns
__global__ __launch_bounds__(256)
void cvt_pad_k(const float* __restrict__ s, _Float16* __restrict__ d,
               int rows, int nc, int np) {
  size_t i = (size_t)blockIdx.x * 256 + threadIdx.x;
  if (i >= (size_t)rows * np) return;
  int n = (int)(i % np);
  int r = (int)(i / np);
  d[i] = (n < nc) ? (_Float16)s[(size_t)r * nc + n] : (_Float16)0.0f;
}

// ---------------- embedding / stream setup ----------------
__global__ __launch_bounds__(256)
void embed_k(const int* __restrict__ qd, const int* __restrict__ tg,
             const float* __restrict__ pe, const float* __restrict__ qemb,
             const float* __restrict__ qaemb,
             float* __restrict__ x32, _Float16* __restrict__ x16,
             _Float16* __restrict__ y16, _Float16* __restrict__ qe16) {
  size_t i = (size_t)blockIdx.x * 256 + threadIdx.x;
  if (i >= (size_t)Mz * Dz) return;
  int d   = (int)(i % Dz);
  int tok = (int)(i / Dz);
  int s   = tok % Sz;
  float qe = qemb[(size_t)qd[tok] * Dz + d];
  float p  = pe[(size_t)s * Dz + d];
  float xv = qe + p;
  float yv = qaemb[(size_t)tg[tok] * Dz + d] + xv;
  x32[i]  = xv;
  x16[i]  = (_Float16)xv;
  y16[i]  = (_Float16)yv;
  qe16[i] = (_Float16)qe;
}

// ---------------- concat [x | qe] ----------------
__global__ __launch_bounds__(256)
void concat_k(const _Float16* __restrict__ a, const _Float16* __restrict__ b,
              _Float16* __restrict__ o) {
  size_t i = (size_t)blockIdx.x * 256 + threadIdx.x;
  if (i >= (size_t)Mz * 2 * Dz) return;
  int    c   = (int)(i % (2 * Dz));
  size_t tok = i / (2 * Dz);
  o[i] = (c < Dz) ? a[tok * Dz + c] : b[tok * Dz + (c - Dz)];
}

// ---------------- generic f16 WMMA GEMM ----------------
// C[M,Nc] = A[M,K] * B[K,Nb(>=Nc, mult of 8)] + bias, optional relu.
// 256 threads = 8 waves; block tile 128x64; wave tile 32x32 (2x2 frags); BK=32.
// Async double-buffered LDS staging on CDNA5 (GLOBAL_LOAD_ASYNC_TO_LDS_B128).
__global__ __launch_bounds__(256)
void gemm_f16(const _Float16* __restrict__ A, const _Float16* __restrict__ Bw,
              const float* __restrict__ bias, float* __restrict__ Cf,
              _Float16* __restrict__ Ch, int M, int Nc, int Nb, int K,
              int relu) {
#if HAS_ASYNC_LDS
  __shared__ _Float16 As[2][128][40];
  __shared__ _Float16 Bs[2][32][72];
#else
  __shared__ _Float16 As[1][128][40];
  __shared__ _Float16 Bs[1][32][72];
#endif

  const int tid  = threadIdx.x;
  const int lane = tid & 31;
  const int wave = tid >> 5;
  const int wm   = wave >> 1;      // 0..3
  const int wn   = wave & 1;       // 0..1
  const int hlf  = lane >> 4;      // wave32 lane halves
  const int id16 = lane & 15;
  const int m0 = blockIdx.y * 128;
  const int n0 = blockIdx.x * 64;

  const int ar = tid >> 1;          // 0..127
  const int ac = (tid & 1) << 4;    // 0 or 16
  const int br = tid >> 3;          // 0..31
  const int bc = (tid & 7) << 3;    // 0..56

  v8f acc[2][2] = {};

  const _Float16* aptr = A + (size_t)(m0 + ar) * K + ac;
  const _Float16* bptr = Bw + (size_t)br * Nb + n0 + bc;

  auto compute = [&](int buf) {
    v16h afr[2], bfr[2];
#pragma unroll
    for (int sm = 0; sm < 2; ++sm) {
      int row = wm * 32 + sm * 16 + id16;
#pragma unroll
      for (int g = 0; g < 2; ++g)
#pragma unroll
        for (int v = 0; v < 4; ++v) {
          int kk = g * 16 + hlf * 8 + v * 2;
          afr[sm][(g * 4 + v) * 2]     = As[buf][row][kk];
          afr[sm][(g * 4 + v) * 2 + 1] = As[buf][row][kk + 1];
        }
    }
#pragma unroll
    for (int sn = 0; sn < 2; ++sn) {
      int col = wn * 32 + sn * 16 + id16;
#pragma unroll
      for (int v = 0; v < 8; ++v) {
        int kk = (v >> 2) * 16 + hlf * 8 + (v & 3) * 2;
        bfr[sn][2 * v]     = Bs[buf][kk][col];
        bfr[sn][2 * v + 1] = Bs[buf][kk + 1][col];
      }
    }
#pragma unroll
    for (int sm = 0; sm < 2; ++sm)
#pragma unroll
      for (int sn = 0; sn < 2; ++sn)
        acc[sm][sn] = wmma_f32_f16(afr[sm], bfr[sn], acc[sm][sn]);
  };

#if HAS_ASYNC_LDS
  auto issue = [&](int buf, int k0) {
    __builtin_amdgcn_global_load_async_to_lds_b128(
        GPTR128(aptr + k0), LPTR128(&As[buf][ar][ac]), 0, 0);
    __builtin_amdgcn_global_load_async_to_lds_b128(
        GPTR128(aptr + k0 + 8), LPTR128(&As[buf][ar][ac + 8]), 0, 0);
    __builtin_amdgcn_global_load_async_to_lds_b128(
        GPTR128(bptr + (size_t)k0 * Nb), LPTR128(&Bs[buf][br][bc]), 0, 0);
  };
  issue(0, 0);
  int buf = 0;
  for (int k0 = 0; k0 < K; k0 += 32, buf ^= 1) {
    wait_async0();     // this wave's async tile landed ...
    __syncthreads();   // ... and every other wave's too
    if (k0 + 32 < K) issue(buf ^ 1, k0 + 32);
    compute(buf);
  }
#else
  for (int k0 = 0; k0 < K; k0 += 32) {
    if (k0 + 32 < K) {  // gfx1250 global_prefetch_b8 for the next K-tile
      __builtin_prefetch(aptr + k0 + 32, 0, 1);
      __builtin_prefetch(bptr + (size_t)(k0 + 32) * Nb, 0, 1);
    }
    uint4 a0 = *(const uint4*)(aptr + k0);
    uint4 a1 = *(const uint4*)(aptr + k0 + 8);
    uint4 b0 = *(const uint4*)(bptr + (size_t)k0 * Nb);
    __syncthreads();
    *(uint4*)&As[0][ar][ac]     = a0;
    *(uint4*)&As[0][ar][ac + 8] = a1;
    *(uint4*)&Bs[0][br][bc]     = b0;
    __syncthreads();
    compute(0);
  }
#endif

  // epilogue: C layout M = v + 8*hlf, N = id16
#pragma unroll
  for (int sm = 0; sm < 2; ++sm) {
    int row0 = m0 + wm * 32 + sm * 16 + hlf * 8;
#pragma unroll
    for (int sn = 0; sn < 2; ++sn) {
      int col = n0 + wn * 32 + sn * 16 + id16;
      if (col < Nc) {
        float bb = bias ? bias[col] : 0.0f;
#pragma unroll
        for (int v = 0; v < 8; ++v) {
          float val = acc[sm][sn][v] + bb;
          if (relu) val = fmaxf(val, 0.0f);
          size_t idx = (size_t)(row0 + v) * Nc + col;
          if (Cf) Cf[idx] = val;
          if (Ch) Ch[idx] = (_Float16)val;
        }
      }
    }
  }
}

// ---------------- flash attention (causal j<i, row0 zeroed) ----------------
// grid = B*H*(S/128); block = 256 (8 waves, 16 query rows each).
__global__ __launch_bounds__(256)
void attn_k(const _Float16* __restrict__ KQ, const _Float16* __restrict__ Vg,
            _Float16* __restrict__ O) {
  __shared__ _Float16 Qs[128][72];
  __shared__ _Float16 Kc[32][72];
  __shared__ _Float16 Vc[32][72];
  __shared__ _Float16 Ps[8][16][32];  // per-wave P (C-layout -> A-layout)

  const int tid  = threadIdx.x;
  const int lane = tid & 31;
  const int wave = tid >> 5;
  const int hlf  = lane >> 4;
  const int id16 = lane & 15;

  const int nqb  = Sz / 128;
  int blk  = blockIdx.x;
  int qblk = blk % nqb;
  int bh   = blk / nqb;
  int b    = bh / Hz, h = bh % Hz;
  int s0b  = qblk * 128;
  int s0   = s0b + wave * 16;

  // stage the 128-row Q block (Q == K projection: kq_same=1)
  for (int i = tid; i < 128 * (DKz / 8); i += 256) {
    int r = i >> 3, c = (i & 7) << 3;
    const _Float16* src = KQ + (size_t)(b * Sz + s0b + r) * Dz + h * DKz + c;
#if HAS_ASYNC_LDS
    __builtin_amdgcn_global_load_async_to_lds_b128(GPTR128(src),
                                                   LPTR128(&Qs[r][c]), 0, 0);
#else
    *(uint4*)&Qs[r][c] = *(const uint4*)src;
#endif
  }
  wait_async0();
  __syncthreads();

  // Q fragments: A-layout, two K=32 chunks covering DK=64
  v16h aq[2];
#pragma unroll
  for (int cch = 0; cch < 2; ++cch)
#pragma unroll
    for (int g = 0; g < 2; ++g)
#pragma unroll
      for (int v = 0; v < 4; ++v) {
        int kk = cch * 32 + g * 16 + hlf * 8 + v * 2;
        aq[cch][(g * 4 + v) * 2]     = Qs[wave * 16 + id16][kk];
        aq[cch][(g * 4 + v) * 2 + 1] = Qs[wave * 16 + id16][kk + 1];
      }

  v8f   acc[4] = {};
  float mrow[8], lrow[8];
#pragma unroll
  for (int v = 0; v < 8; ++v) { mrow[v] = -1e30f; lrow[v] = 0.0f; }

  const int kmax = s0b + 128;  // keys strictly < last query row + 1
  for (int kb = 0; kb < kmax; kb += 32) {
    __syncthreads();
    {  // stage 32 K rows and 32 V rows
      int r = tid >> 3, c = (tid & 7) << 3;
      size_t gk = (size_t)(b * Sz + kb + r) * Dz + h * DKz + c;
#if HAS_ASYNC_LDS
      __builtin_amdgcn_global_load_async_to_lds_b128(GPTR128(KQ + gk),
                                                     LPTR128(&Kc[r][c]), 0, 0);
      __builtin_amdgcn_global_load_async_to_lds_b128(GPTR128(Vg + gk),
                                                     LPTR128(&Vc[r][c]), 0, 0);
      wait_async0();
#else
      *(uint4*)&Kc[r][c] = *(const uint4*)(KQ + gk);
      *(uint4*)&Vc[r][c] = *(const uint4*)(Vg + gk);
#endif
    }
    __syncthreads();

    // scores: two 16-key tiles, contraction over DK=64 (2 WMMAs each)
    v8f sc0 = {}, sc1 = {};
#pragma unroll
    for (int t = 0; t < 2; ++t) {
      v8f s_ = {};
#pragma unroll
      for (int cch = 0; cch < 2; ++cch) {
        v16h bk_;
#pragma unroll
        for (int v = 0; v < 8; ++v) {
          int kk = cch * 32 + (v >> 2) * 16 + hlf * 8 + (v & 3) * 2;
          bk_[2 * v]     = Kc[t * 16 + id16][kk];
          bk_[2 * v + 1] = Kc[t * 16 + id16][kk + 1];
        }
        s_ = wmma_f32_f16(aq[cch], bk_, s_);
      }
      if (t == 0) sc0 = s_; else sc1 = s_;
    }

    // online softmax (row = v + 8*hlf spread over a 16-lane group)
    const float scl = 0.125f;  // 1/sqrt(64)
#pragma unroll
    for (int v = 0; v < 8; ++v) {
      int   sq = s0 + v + hlf * 8;
      float x0 = (kb + id16 < sq)      ? sc0[v] * scl : -1e30f;
      float x1 = (kb + 16 + id16 < sq) ? sc1[v] * scl : -1e30f;
      float bm = fmaxf(x0, x1);
      bm = fmaxf(bm, __shfl_xor(bm, 1, 32));
      bm = fmaxf(bm, __shfl_xor(bm, 2, 32));
      bm = fmaxf(bm, __shfl_xor(bm, 4, 32));
      bm = fmaxf(bm, __shfl_xor(bm, 8, 32));
      float mn = fmaxf(mrow[v], bm);
      float co = __expf(mrow[v] - mn);
      float p0 = __expf(x0 - mn);
      float p1 = __expf(x1 - mn);
      float rs = p0 + p1;
      rs += __shfl_xor(rs, 1, 32);
      rs += __shfl_xor(rs, 2, 32);
      rs += __shfl_xor(rs, 4, 32);
      rs += __shfl_xor(rs, 8, 32);
      lrow[v] = lrow[v] * co + rs;
      mrow[v] = mn;
      acc[0][v] *= co; acc[1][v] *= co; acc[2][v] *= co; acc[3][v] *= co;
      Ps[wave][v + hlf * 8][id16]      = (_Float16)p0;
      Ps[wave][v + hlf * 8][16 + id16] = (_Float16)p1;
    }

    // P (C-layout) -> A-layout via per-wave LDS (DS ops in-order per wave)
    v16h ap;
#pragma unroll
    for (int g = 0; g < 2; ++g)
#pragma unroll
      for (int v = 0; v < 4; ++v) {
        int kk = g * 16 + hlf * 8 + v * 2;
        ap[(g * 4 + v) * 2]     = Ps[wave][id16][kk];
        ap[(g * 4 + v) * 2 + 1] = Ps[wave][id16][kk + 1];
      }
    // PV: P(16x32) * V(32x64) -> 4 x 16x16 accumulators
#pragma unroll
    for (int d = 0; d < 4; ++d) {
      v16h bv_;
#pragma unroll
      for (int v = 0; v < 8; ++v) {
        int kk = (v >> 2) * 16 + hlf * 8 + (v & 3) * 2;
        bv_[2 * v]     = Vc[kk][d * 16 + id16];
        bv_[2 * v + 1] = Vc[kk + 1][d * 16 + id16];
      }
      acc[d] = wmma_f32_f16(ap, bv_, acc[d]);
    }
  }

  // epilogue: normalize; zero row s==0 (reference's row_keep)
#pragma unroll
  for (int d = 0; d < 4; ++d)
#pragma unroll
    for (int v = 0; v < 8; ++v) {
      int   sq = s0 + v + hlf * 8;
      float ov = (sq == 0) ? 0.0f : acc[d][v] / lrow[v];
      O[(size_t)(b * Sz + sq) * Dz + h * DKz + d * 16 + id16] = (_Float16)ov;
    }
}

// ---------------- fused residual-add + LayerNorm (1 wave per row) ----------------
__global__ __launch_bounds__(256)
void add_ln(const float* __restrict__ xin, const float* __restrict__ res,
            const float* __restrict__ g, const float* __restrict__ bta,
            float* __restrict__ xout, _Float16* __restrict__ xh) {
  int row  = blockIdx.x * 8 + (threadIdx.x >> 5);
  int lane = threadIdx.x & 31;
  size_t base = (size_t)row * Dz;
  float v[16], s = 0.0f;
#pragma unroll
  for (int i = 0; i < 16; ++i) {
    int c = lane + i * 32;
    v[i] = xin[base + c] + res[base + c];
    s += v[i];
  }
#pragma unroll
  for (int m = 1; m < 32; m <<= 1) s += __shfl_xor(s, m, 32);
  float mean = s * (1.0f / Dz);
  float var = 0.0f;
#pragma unroll
  for (int i = 0; i < 16; ++i) { float d = v[i] - mean; var += d * d; }
#pragma unroll
  for (int m = 1; m < 32; m <<= 1) var += __shfl_xor(var, m, 32);
  float rstd = rsqrtf(var * (1.0f / Dz) + 1e-5f);
#pragma unroll
  for (int i = 0; i < 16; ++i) {
    int c = lane + i * 32;
    float o = (v[i] - mean) * rstd * g[c] + bta[c];
    xout[base + c] = o;
    xh[base + c]   = (_Float16)o;
  }
}

// ---------------- host orchestration ----------------
extern "C" void kernel_launch(void* const* d_in, const int* in_sizes, int n_in,
                              void* d_out, int out_size, void* d_ws, size_t ws_size,
                              hipStream_t stream) {
  (void)in_sizes; (void)n_in; (void)out_size; (void)ws_size;
  const int*   qd    = (const int*)d_in[0];
  const int*   tg    = (const int*)d_in[1];
  const float* pe    = (const float*)d_in[2];
  const float* qemb  = (const float*)d_in[3];
  const float* qaemb = (const float*)d_in[4];
  const float* Wk = (const float*)d_in[5];  const float* bk = (const float*)d_in[6];
  const float* Wv = (const float*)d_in[7];  const float* bv = (const float*)d_in[8];
  const float* Wo = (const float*)d_in[9];  const float* bo = (const float*)d_in[10];
  const float* ln1g = (const float*)d_in[11]; const float* ln1b = (const float*)d_in[12];
  const float* W1 = (const float*)d_in[13]; const float* b1 = (const float*)d_in[14];
  const float* W2 = (const float*)d_in[15]; const float* b2 = (const float*)d_in[16];
  const float* ln2g = (const float*)d_in[17]; const float* ln2b = (const float*)d_in[18];
  const float* Wh1 = (const float*)d_in[19]; const float* bh1 = (const float*)d_in[20];
  const float* Wh2 = (const float*)d_in[21]; const float* bh2 = (const float*)d_in[22];
  const float* Wh3 = (const float*)d_in[23]; const float* bh3 = (const float*)d_in[24];

  char*  ws  = (char*)d_ws;
  size_t off = 0;
  auto alloc = [&](size_t bytes) -> char* {
    char* p = ws + off;
    off += (bytes + 255) & ~(size_t)255;
    return p;
  };
  float*    x32   = (float*)alloc((size_t)Mz * Dz * 4);
  _Float16* x16   = (_Float16*)alloc((size_t)Mz * Dz * 2);
  _Float16* y16   = (_Float16*)alloc((size_t)Mz * Dz * 2);
  _Float16* qe16  = (_Float16*)alloc((size_t)Mz * Dz * 2);
  _Float16* kq16  = (_Float16*)alloc((size_t)Mz * Dz * 2);   // reused as head h1
  _Float16* v16b  = (_Float16*)alloc((size_t)Mz * Dz * 2);   // reused as head h2
  _Float16* at16  = (_Float16*)alloc((size_t)Mz * Dz * 2);
  float*    of32  = (float*)alloc((size_t)Mz * Dz * 4);
  _Float16* big16 = (_Float16*)alloc((size_t)Mz * DFFz * 2); // FFN hidden / concat
  _Float16* wk16  = (_Float16*)alloc((size_t)Lz * Dz * Dz * 2);
  _Float16* wv16  = (_Float16*)alloc((size_t)Lz * Dz * Dz * 2);
  _Float16* wo16  = (_Float16*)alloc((size_t)Lz * Dz * Dz * 2);
  _Float16* w116  = (_Float16*)alloc((size_t)Lz * Dz * DFFz * 2);
  _Float16* w216  = (_Float16*)alloc((size_t)Lz * DFFz * Dz * 2);
  _Float16* wh1h  = (_Float16*)alloc((size_t)2 * Dz * FC1z * 2);
  _Float16* wh2h  = (_Float16*)alloc((size_t)FC1z * FC2z * 2);
  _Float16* wh3h  = (_Float16*)alloc((size_t)FC2z * NSKp * 2);  // padded

  auto cvt = [&](const float* s, _Float16* d, size_t n) {
    cvt_k<<<(unsigned)((n + 255) / 256), 256, 0, stream>>>(s, d, n);
  };
  auto gemm = [&](const _Float16* A, const _Float16* Bw, const float* bias,
                  float* Cf, _Float16* Ch, int M_, int Nc_, int Nb_, int K_,
                  int relu) {
    dim3 g((Nc_ + 63) / 64, M_ / 128);
    gemm_f16<<<g, 256, 0, stream>>>(A, Bw, bias, Cf, Ch, M_, Nc_, Nb_, K_, relu);
  };

  cvt(Wk, wk16, (size_t)Lz * Dz * Dz);
  cvt(Wv, wv16, (size_t)Lz * Dz * Dz);
  cvt(Wo, wo16, (size_t)Lz * Dz * Dz);
  cvt(W1, w116, (size_t)Lz * Dz * DFFz);
  cvt(W2, w216, (size_t)Lz * DFFz * Dz);
  cvt(Wh1, wh1h, (size_t)2 * Dz * FC1z);
  cvt(Wh2, wh2h, (size_t)FC1z * FC2z);
  {
    size_t n = (size_t)FC2z * NSKp;
    cvt_pad_k<<<(unsigned)((n + 255) / 256), 256, 0, stream>>>(Wh3, wh3h, FC2z,
                                                               NSKz, NSKp);
  }

  {
    size_t n = (size_t)Mz * Dz;
    embed_k<<<(unsigned)((n + 255) / 256), 256, 0, stream>>>(
        qd, tg, pe, qemb, qaemb, x32, x16, y16, qe16);
  }

  for (int l = 0; l < Lz; ++l) {
    gemm(x16, wk16 + (size_t)l * Dz * Dz, bk + l * Dz, nullptr, kq16, Mz, Dz, Dz, Dz, 0);
    gemm(y16, wv16 + (size_t)l * Dz * Dz, bv + l * Dz, nullptr, v16b, Mz, Dz, Dz, Dz, 0);
    attn_k<<<Bz * Hz * (Sz / 128), 256, 0, stream>>>(kq16, v16b, at16);
    gemm(at16, wo16 + (size_t)l * Dz * Dz, bo + l * Dz, of32, nullptr, Mz, Dz, Dz, Dz, 0);
    add_ln<<<Mz / 8, 256, 0, stream>>>(x32, of32, ln1g + l * Dz, ln1b + l * Dz, x32, x16);
    gemm(x16, w116 + (size_t)l * Dz * DFFz, b1 + l * DFFz, nullptr, big16, Mz, DFFz, DFFz, Dz, 1);
    gemm(big16, w216 + (size_t)l * DFFz * Dz, b2 + l * Dz, of32, nullptr, Mz, Dz, Dz, DFFz, 0);
    add_ln<<<Mz / 8, 256, 0, stream>>>(x32, of32, ln2g + l * Dz, ln2b + l * Dz, x32, x16);
  }

  {
    size_t n = (size_t)Mz * 2 * Dz;
    concat_k<<<(unsigned)((n + 255) / 256), 256, 0, stream>>>(x16, qe16, big16);
  }
  gemm(big16, wh1h, bh1, nullptr, kq16, Mz, FC1z, FC1z, 2 * Dz, 1);
  gemm(kq16, wh2h, bh2, nullptr, v16b, Mz, FC2z, FC2z, FC1z, 1);
  gemm(v16b, wh3h, bh3, (float*)d_out, nullptr, Mz, NSKz, NSKp, FC2z, 0);
}
